// RNNVAE_70231305224765
// MI455X (gfx1250) — compile-verified, matrix operand order checked by reading, not run on hardware
//
#include <hip/hip_runtime.h>
#include <hip/hip_bf16.h>

typedef __attribute__((ext_vector_type(16))) _Float16 v16h;
typedef __attribute__((ext_vector_type(8)))  float    v8f;

#define B_   64
#define T_   1024
#define D_   64
#define H_   256
#define G_   768   // 3H
#define L_   64
#define O_   64

// Map matrix element (row m, col k) to half-index in packed WMMA fragment layout.
// Per the CDNA5 16-bit A-matrix layout: lanes 0-15 hold K 0..7 / 16..23
// (e=0..7 / 8..15), lanes 16-31 hold K 8..15 / 24..31.  ntk = K/32.
__device__ __forceinline__ int frag_pos(int m, int k, int ntk) {
  int mt = m >> 4, lm = m & 15;
  int kb = k >> 5, kk = k & 31;
  int lane = lm | (((kk >> 3) & 1) << 4);
  int e    = ((kk >> 4) << 3) | (kk & 7);
  return ((mt * ntk + kb) * 32 + lane) * 16 + e;
}

__device__ __forceinline__ float sigmoidf_(float x) {
  return 1.0f / (1.0f + __expf(-x));
}

// Convert fp32 weight matrix (N rows x K cols, row-major) into packed f16
// WMMA-fragment layout (used as B matrices, one output column per lane).
__global__ void pack_f16(const float* __restrict__ src, _Float16* __restrict__ dst,
                         int N, int K) {
  int ntk = K >> 5;
  int total = N * K;
  for (int i = blockIdx.x * blockDim.x + threadIdx.x; i < total;
       i += gridDim.x * blockDim.x) {
    int n = i / K, k = i - n * K;
    dst[frag_pos(n, k, ntk)] = (_Float16)src[i];
  }
}

// ---------------- Encoder: one workgroup per direction ----------------
__global__ __launch_bounds__(1024) void enc_gru(
    const float* __restrict__ x,
    const float* __restrict__ bih_f, const float* __restrict__ bhh_f,
    const float* __restrict__ bih_b, const float* __restrict__ bhh_b,
    const _Float16* __restrict__ WihF, const _Float16* __restrict__ WhhF,
    const _Float16* __restrict__ WihB, const _Float16* __restrict__ WhhB,
    float* __restrict__ h_final) {
  // double-buffered h (packed frag layout, ntk=8) and x_t (ntk=2)
  __shared__ _Float16 ldsH[2][B_ * H_];
  __shared__ _Float16 ldsX[2][B_ * D_];

  const int dir = blockIdx.x;
  const float* __restrict__ bih = dir ? bih_b : bih_f;
  const float* __restrict__ bhh = dir ? bhh_b : bhh_f;
  const _Float16* __restrict__ Wih = dir ? WihB : WihF;
  const _Float16* __restrict__ Whh = dir ? WhhB : WhhF;

  const int tid = threadIdx.x, lane = tid & 31, wave = tid >> 5;
  const int mt = wave & 3;       // M tile (batch rows)
  const int cb0 = wave >> 2;     // first column block (of H); second is cb0+8

  for (int i = tid; i < B_ * H_; i += 1024) ldsH[0][i] = (_Float16)0.0f;

  for (int t = 0; t < T_; ++t) {
    const int cur = t & 1, nxt = cur ^ 1;
    const int tt = dir ? (T_ - 1 - t) : t;

    // stage x_t (64x64) into packed fragment layout (own parity buffer)
    for (int i = tid; i < B_ * D_; i += 1024) {
      int m = i >> 6, k = i & 63;
      ldsX[cur][frag_pos(m, k, 2)] = (_Float16)x[(m * T_ + tt) * D_ + k];
    }
    __syncthreads();   // staging visible; also fences prev step's h writes

    const _Float16* Hc = ldsH[cur];
    const _Float16* Xc = ldsX[cur];
    _Float16*       Hn = ldsH[nxt];

#pragma clang loop unroll(disable)
    for (int p = 0; p < 2; ++p) {
      int cb = cb0 + p * 8;
      v8f ar = {}, az = {}, ai = {}, ah = {};
      // input-gate GEMM: gi = x_t @ Wih^T  (K = 64)
#pragma unroll
      for (int kb = 0; kb < 2; ++kb) {
        v16h a  = *(const v16h*)(Xc + ((mt * 2 + kb) * 32 + lane) * 16);
        v16h br = *(const v16h*)(Wih + (((0 * 16 + cb) * 2 + kb) * 32 + lane) * 16);
        v16h bz = *(const v16h*)(Wih + (((1 * 16 + cb) * 2 + kb) * 32 + lane) * 16);
        v16h bn = *(const v16h*)(Wih + (((2 * 16 + cb) * 2 + kb) * 32 + lane) * 16);
        ar = __builtin_amdgcn_wmma_f32_16x16x32_f16(false, a, false, br, (short)0, ar, false, false);
        az = __builtin_amdgcn_wmma_f32_16x16x32_f16(false, a, false, bz, (short)0, az, false, false);
        ai = __builtin_amdgcn_wmma_f32_16x16x32_f16(false, a, false, bn, (short)0, ai, false, false);
      }
      // hidden-gate GEMM: gh = h @ Whh^T  (K = 256)
#pragma unroll
      for (int kb = 0; kb < 8; ++kb) {
        v16h a  = *(const v16h*)(Hc + ((mt * 8 + kb) * 32 + lane) * 16);
        v16h br = *(const v16h*)(Whh + (((0 * 16 + cb) * 8 + kb) * 32 + lane) * 16);
        v16h bz = *(const v16h*)(Whh + (((1 * 16 + cb) * 8 + kb) * 32 + lane) * 16);
        v16h bn = *(const v16h*)(Whh + (((2 * 16 + cb) * 8 + kb) * 32 + lane) * 16);
        ar = __builtin_amdgcn_wmma_f32_16x16x32_f16(false, a, false, br, (short)0, ar, false, false);
        az = __builtin_amdgcn_wmma_f32_16x16x32_f16(false, a, false, bz, (short)0, az, false, false);
        ah = __builtin_amdgcn_wmma_f32_16x16x32_f16(false, a, false, bn, (short)0, ah, false, false);
      }
      // GRU combine; write new h into the other parity buffer
#pragma unroll
      for (int v = 0; v < 8; ++v) {
        int m   = mt * 16 + v + ((lane >> 4) << 3);
        int col = cb * 16 + (lane & 15);
        float hold = (float)Hc[frag_pos(m, col, 8)];
        float r  = sigmoidf_(ar[v] + bih[col] + bhh[col]);
        float zz = sigmoidf_(az[v] + bih[256 + col] + bhh[256 + col]);
        float nn = tanhf(ai[v] + bih[512 + col] + r * (ah[v] + bhh[512 + col]));
        Hn[frag_pos(m, col, 8)] = (_Float16)((1.0f - zz) * nn + zz * hold);
      }
    }
  }
  __syncthreads();

  // emit final hidden state: fwd -> cols [0,256), bwd -> cols [256,512)
  const _Float16* Hf = ldsH[T_ & 1];
  for (int i = tid; i < B_ * H_; i += 1024) {
    int m = i >> 8, k = i & 255;
    h_final[m * (2 * H_) + dir * H_ + k] = (float)Hf[frag_pos(m, k, 8)];
  }
}

// ---------------- Latent heads + reparameterize (tiny, scalar) ----------------
__global__ void latent_k(const float* __restrict__ hfin, const float* __restrict__ eps,
                         const float* __restrict__ Wmu, const float* __restrict__ bmu,
                         const float* __restrict__ Wlv, const float* __restrict__ blv,
                         float* __restrict__ mu, float* __restrict__ lv,
                         float* __restrict__ z) {
  int i = blockIdx.x * blockDim.x + threadIdx.x;
  if (i >= B_ * L_) return;
  int b = i >> 6, l = i & 63;
  float sm = bmu[l], sv = blv[l];
  for (int k = 0; k < 2 * H_; ++k) {
    float h = hfin[b * (2 * H_) + k];
    sm += h * Wmu[l * (2 * H_) + k];
    sv += h * Wlv[l * (2 * H_) + k];
  }
  mu[i] = sm;
  lv[i] = sv;
  z[i] = sm + eps[i] * __expf(0.5f * sv);
}

__global__ void dec_init_k(const float* __restrict__ z, const float* __restrict__ Wl2h,
                           const float* __restrict__ bl2h, float* __restrict__ h0) {
  int i = blockIdx.x * blockDim.x + threadIdx.x;
  if (i >= B_ * H_) return;
  int b = i >> 8, h = i & 255;
  float s = bl2h[h];
  for (int l = 0; l < L_; ++l) s += z[b * L_ + l] * Wl2h[h * L_ + l];
  h0[i] = tanhf(s);
}

// ---------------- Decoder: single workgroup, autoregressive ----------------
__global__ __launch_bounds__(1024) void dec_gru(
    const float* __restrict__ h0,
    const float* __restrict__ bih, const float* __restrict__ bhh,
    const float* __restrict__ bout,
    const _Float16* __restrict__ Wih, const _Float16* __restrict__ Whh,
    const _Float16* __restrict__ Wout,
    float* __restrict__ xrec) {
  __shared__ _Float16 ldsH[2][B_ * H_];  // h, packed (ntk = 8), double-buffered
  __shared__ _Float16 ldsI[2][B_ * O_];  // feedback input, packed (ntk = 2)

  const int tid = threadIdx.x, lane = tid & 31, wave = tid >> 5;
  const int mt = wave & 3, cb0 = wave >> 2;

  for (int i = tid; i < B_ * H_; i += 1024) {
    int m = i >> 8, k = i & 255;
    ldsH[0][frag_pos(m, k, 8)] = (_Float16)h0[i];
  }
  for (int i = tid; i < B_ * O_; i += 1024) ldsI[0][i] = (_Float16)0.0f;
  __syncthreads();

  for (int t = 0; t < T_; ++t) {
    const int cur = t & 1, nxt = cur ^ 1;
    const _Float16* Hc = ldsH[cur];
    const _Float16* Ic = ldsI[cur];
    _Float16*       Hn = ldsH[nxt];

#pragma clang loop unroll(disable)
    for (int p = 0; p < 2; ++p) {
      int cb = cb0 + p * 8;
      v8f ar = {}, az = {}, ai = {}, ah = {};
#pragma unroll
      for (int kb = 0; kb < 2; ++kb) {
        v16h a  = *(const v16h*)(Ic + ((mt * 2 + kb) * 32 + lane) * 16);
        v16h br = *(const v16h*)(Wih + (((0 * 16 + cb) * 2 + kb) * 32 + lane) * 16);
        v16h bz = *(const v16h*)(Wih + (((1 * 16 + cb) * 2 + kb) * 32 + lane) * 16);
        v16h bn = *(const v16h*)(Wih + (((2 * 16 + cb) * 2 + kb) * 32 + lane) * 16);
        ar = __builtin_amdgcn_wmma_f32_16x16x32_f16(false, a, false, br, (short)0, ar, false, false);
        az = __builtin_amdgcn_wmma_f32_16x16x32_f16(false, a, false, bz, (short)0, az, false, false);
        ai = __builtin_amdgcn_wmma_f32_16x16x32_f16(false, a, false, bn, (short)0, ai, false, false);
      }
#pragma unroll
      for (int kb = 0; kb < 8; ++kb) {
        v16h a  = *(const v16h*)(Hc + ((mt * 8 + kb) * 32 + lane) * 16);
        v16h br = *(const v16h*)(Whh + (((0 * 16 + cb) * 8 + kb) * 32 + lane) * 16);
        v16h bz = *(const v16h*)(Whh + (((1 * 16 + cb) * 8 + kb) * 32 + lane) * 16);
        v16h bn = *(const v16h*)(Whh + (((2 * 16 + cb) * 8 + kb) * 32 + lane) * 16);
        ar = __builtin_amdgcn_wmma_f32_16x16x32_f16(false, a, false, br, (short)0, ar, false, false);
        az = __builtin_amdgcn_wmma_f32_16x16x32_f16(false, a, false, bz, (short)0, az, false, false);
        ah = __builtin_amdgcn_wmma_f32_16x16x32_f16(false, a, false, bn, (short)0, ah, false, false);
      }
#pragma unroll
      for (int v = 0; v < 8; ++v) {
        int m   = mt * 16 + v + ((lane >> 4) << 3);
        int col = cb * 16 + (lane & 15);
        float hold = (float)Hc[frag_pos(m, col, 8)];
        float r  = sigmoidf_(ar[v] + bih[col] + bhh[col]);
        float zz = sigmoidf_(az[v] + bih[256 + col] + bhh[256 + col]);
        float nn = tanhf(ai[v] + bih[512 + col] + r * (ah[v] + bhh[512 + col]));
        Hn[frag_pos(m, col, 8)] = (_Float16)((1.0f - zz) * nn + zz * hold);
      }
    }
    __syncthreads();   // h_new complete

    // output projection: out = h_new @ Wout^T + bout  (64x64, 16 tiles)
    if (wave < 16) {                    // wave-uniform branch: EXEC all-ones per wave
      const int omt = wave & 3, ont = wave >> 2;
      v8f acc = {};
#pragma unroll
      for (int kb = 0; kb < 8; ++kb) {
        v16h a = *(const v16h*)(Hn + ((omt * 8 + kb) * 32 + lane) * 16);
        v16h b = *(const v16h*)(Wout + ((ont * 8 + kb) * 32 + lane) * 16);
        acc = __builtin_amdgcn_wmma_f32_16x16x32_f16(false, a, false, b, (short)0, acc, false, false);
      }
#pragma unroll
      for (int v = 0; v < 8; ++v) {
        int m = omt * 16 + v + ((lane >> 4) << 3);
        int o = ont * 16 + (lane & 15);
        float val = acc[v] + bout[o];
        xrec[(m * T_ + t) * O_ + o] = val;
        ldsI[nxt][frag_pos(m, o, 2)] = (_Float16)val;  // next-step feedback input
      }
    }
    __syncthreads();   // feedback input visible before next step
  }
}

extern "C" void kernel_launch(void* const* d_in, const int* in_sizes, int n_in,
                              void* d_out, int out_size, void* d_ws, size_t ws_size,
                              hipStream_t stream) {
  (void)in_sizes; (void)n_in; (void)out_size; (void)ws_size;

  const float* x        = (const float*)d_in[0];
  const float* eps      = (const float*)d_in[1];
  const float* eWihF    = (const float*)d_in[2];
  const float* eWhhF    = (const float*)d_in[3];
  const float* ebihF    = (const float*)d_in[4];
  const float* ebhhF    = (const float*)d_in[5];
  const float* eWihB    = (const float*)d_in[6];
  const float* eWhhB    = (const float*)d_in[7];
  const float* ebihB    = (const float*)d_in[8];
  const float* ebhhB    = (const float*)d_in[9];
  const float* Wmu      = (const float*)d_in[10];
  const float* bmu      = (const float*)d_in[11];
  const float* Wlv      = (const float*)d_in[12];
  const float* blv      = (const float*)d_in[13];
  const float* Wl2h     = (const float*)d_in[14];
  const float* bl2h     = (const float*)d_in[15];
  const float* dWih     = (const float*)d_in[16];
  const float* dWhh     = (const float*)d_in[17];
  const float* dbih     = (const float*)d_in[18];
  const float* dbhh     = (const float*)d_in[19];
  const float* Woutf    = (const float*)d_in[20];
  const float* bout     = (const float*)d_in[21];

  // workspace layout: packed f16 weights, then fp32 intermediates
  _Float16* wihFp = (_Float16*)d_ws;
  _Float16* whhFp = wihFp + G_ * D_;           // 49152
  _Float16* wihBp = whhFp + G_ * H_;           // 196608
  _Float16* whhBp = wihBp + G_ * D_;
  _Float16* wihDp = whhBp + G_ * H_;
  _Float16* whhDp = wihDp + G_ * D_;
  _Float16* woutP = whhDp + G_ * H_;           // 16384 halves
  float* hfin  = (float*)(woutP + O_ * H_);    // 64 x 512
  float* zbuf  = hfin + B_ * 2 * H_;           // 64 x 64
  float* h0buf = zbuf + B_ * L_;               // 64 x 256

  float* xrec = (float*)d_out;
  float* muO  = xrec + B_ * T_ * O_;
  float* lvO  = muO + B_ * L_;

  // 1) pack weights to f16 WMMA-fragment layout (weights stay hot in L2)
  pack_f16<<<192, 256, 0, stream>>>(eWihF, wihFp, G_, D_);
  pack_f16<<<192, 256, 0, stream>>>(eWhhF, whhFp, G_, H_);
  pack_f16<<<192, 256, 0, stream>>>(eWihB, wihBp, G_, D_);
  pack_f16<<<192, 256, 0, stream>>>(eWhhB, whhBp, G_, H_);
  pack_f16<<<192, 256, 0, stream>>>(dWih,  wihDp, G_, D_);
  pack_f16<<<192, 256, 0, stream>>>(dWhh,  whhDp, G_, H_);
  pack_f16<<<192, 256, 0, stream>>>(Woutf, woutP, O_, H_);

  // 2) bidirectional encoder: block 0 = forward, block 1 = backward
  enc_gru<<<2, 1024, 0, stream>>>(x, ebihF, ebhhF, ebihB, ebhhB,
                                  wihFp, whhFp, wihBp, whhBp, hfin);

  // 3) latent heads + reparameterize, 4) decoder initial hidden
  latent_k<<<(B_ * L_ + 255) / 256, 256, 0, stream>>>(hfin, eps, Wmu, bmu, Wlv, blv,
                                                      muO, lvO, zbuf);
  dec_init_k<<<(B_ * H_ + 255) / 256, 256, 0, stream>>>(zbuf, Wl2h, bl2h, h0buf);

  // 5) autoregressive decoder
  dec_gru<<<1, 1024, 0, stream>>>(h0buf, dbih, dbhh, bout,
                                  wihDp, whhDp, woutP, xrec);
}
